// ResidualHybridGraphGenerator_28948079575765
// MI455X (gfx1250) — compile-verified, matrix operand order checked by reading, not run on hardware
//
#include <hip/hip_runtime.h>
#include <hip/hip_bf16.h>
#include <math.h>

typedef __attribute__((ext_vector_type(16))) _Float16 v16h;
typedef __attribute__((ext_vector_type(8)))  float    v8f;

#define NN   768
#define BB   2
#define TT   96
#define FM   8
#define HID  64

// ---------------------------------------------------------------------------
// Kernel 1: per-node features  -> h (B,N,64), s_i (B,N,64), s_j (B,N,64)
// ---------------------------------------------------------------------------
__global__ __launch_bounds__(64) void node_feat_kernel(
    const float* __restrict__ x_hist,   // B,N,T
    const float* __restrict__ x_mark,   // B,N,T,FM
    const float* __restrict__ te_w1, const float* __restrict__ te_b1,
    const float* __restrict__ te_w2, const float* __restrict__ te_b2,
    const float* __restrict__ me_w1, const float* __restrict__ me_b1,
    const float* __restrict__ me_w2, const float* __restrict__ me_b2,
    const float* __restrict__ nf_w,  const float* __restrict__ nf_b,
    const float* __restrict__ ps_w1,
    float* __restrict__ hg, float* __restrict__ sig, float* __restrict__ sjg)
{
    const int row = blockIdx.x;            // b*N + n
    const int tid = threadIdx.x;

    __shared__ float red[64], redm[64];
    __shared__ float stats3[3], msum[FM];
    __shared__ float t1[32], t2[32], m1[32], m2[32], hrow[64];

    // ---- stats over T: last / mean / max ----
    const float* xh = x_hist + (size_t)row * TT;
    float s = 0.f, mx = -3.4e38f;
    for (int t = tid; t < TT; t += 64) { float v = xh[t]; s += v; mx = fmaxf(mx, v); }
    red[tid] = s; redm[tid] = mx;
    __syncthreads();
    for (int off = 32; off > 0; off >>= 1) {
        if (tid < off) { red[tid] += red[tid + off]; redm[tid] = fmaxf(redm[tid], redm[tid + off]); }
        __syncthreads();
    }
    if (tid == 0) {
        stats3[0] = xh[TT - 1];
        stats3[1] = red[0] * (1.f / (float)TT);
        stats3[2] = redm[0];
    }

    // ---- mark mean over T (8 features) ----
    const float* xm = x_mark + (size_t)row * TT * FM;
    float a = 0.f;
    {
        int f = tid & 7, g = tid >> 3;         // 8 thread-groups per feature
        for (int t = g; t < TT; t += 8) a += xm[t * FM + f];
    }
    __syncthreads();
    red[tid] = a;
    __syncthreads();
    if (tid < FM) {
        float v = 0.f;
        for (int g = 0; g < 8; ++g) v += red[g * FM + tid];
        msum[tid] = v * (1.f / (float)TT);
    }
    __syncthreads();

    // ---- tiny MLPs ----
    if (tid < 32) {
        float acc = te_b1[tid];
        for (int k = 0; k < 3; ++k) acc = fmaf(stats3[k], te_w1[k * 32 + tid], acc);
        t1[tid] = fmaxf(acc, 0.f);
        float accm = me_b1[tid];
        for (int k = 0; k < FM; ++k) accm = fmaf(msum[k], me_w1[k * 32 + tid], accm);
        m1[tid] = fmaxf(accm, 0.f);
    }
    __syncthreads();
    if (tid < 32) {
        float acc = te_b2[tid];
        for (int k = 0; k < 32; ++k) acc = fmaf(t1[k], te_w2[k * 32 + tid], acc);
        t2[tid] = acc;
        float accm = me_b2[tid];
        for (int k = 0; k < 32; ++k) accm = fmaf(m1[k], me_w2[k * 32 + tid], accm);
        m2[tid] = accm;
    }
    __syncthreads();
    {
        float acc = nf_b[tid];
        for (int k = 0; k < 32; ++k) acc = fmaf(t2[k], nf_w[k * HID + tid], acc);
        for (int k = 0; k < 32; ++k) acc = fmaf(m2[k], nf_w[(32 + k) * HID + tid], acc);
        float hv = fmaxf(acc, 0.f);
        hrow[tid] = hv;
        hg[(size_t)row * HID + tid] = hv;
    }
    __syncthreads();
    {
        float ai = 0.f, aj = 0.f;
        for (int k = 0; k < HID; ++k) {
            float hk = hrow[k];
            ai = fmaf(hk, ps_w1[k * HID + tid], ai);               // w_i
            aj = fmaf(hk, ps_w1[(HID + k) * HID + tid], aj);       // w_j
        }
        sig[(size_t)row * HID + tid] = ai;
        sjg[(size_t)row * HID + tid] = aj;
    }
}

// ---------------------------------------------------------------------------
// Kernel 2: pairwise  delta_raw[b,i,j] = tanh(relu(|h_i-h_j|@w_d + s_i + s_j + b1)@w2 + b2)
// One wave32 per 16x16 (i,j) tile; 8 WMMAs per i (4 feature tiles x K=64/32).
// h_j fragment-pattern values and s_j are hoisted to registers, s_i tile is
// staged in LDS, so the i-loop touches only DS (h_i broadcast reads) + WMMA.
// ---------------------------------------------------------------------------
__global__ __launch_bounds__(32) void pair_wmma_kernel(
    const float* __restrict__ hg,
    const float* __restrict__ sig,
    const float* __restrict__ sjg,
    const float* __restrict__ ps_w1,   // w_d = rows 128..191
    const float* __restrict__ ps_b1,
    const float* __restrict__ ps_w2,
    const float* __restrict__ ps_b2,
    float* __restrict__ delta_raw)
{
    const int jt   = blockIdx.x;           // j tile 0..47
    const int it   = blockIdx.y;           // i tile 0..47
    const int b    = blockIdx.z;
    const int lane = threadIdx.x;
    const int lr   = lane & 15;            // M-row / N-col within tile
    const int hi   = lane >> 4;            // lane-half selector

    __shared__ float lhi[16 * HID];        // h rows of the i tile
    __shared__ float lsi[16 * HID];        // s_i rows of the i tile

    const int ibase = it * 16, jbase = jt * 16;
    const float* hb  = hg  + (size_t)b * NN * HID;
    const float* sib = sig + (size_t)b * NN * HID;

    for (int idx = lane; idx < 16 * HID; idx += 32) {
        int r = idx >> 6, c = idx & 63;
        lhi[idx] = hb[(size_t)(ibase + r) * HID + c];
        lsi[idx] = sib[(size_t)(ibase + r) * HID + c];
    }
    __syncthreads();

    // Per-lane K pattern (ISA 16-bit A/B layout):
    //   K(c,t) = c*32 + (hi?8:0) + (t<8 ? t : t+8)
    // h_j row for this lane is lr; its 32 patterned values are i-invariant.
    float hjv[2][16];
#pragma unroll
    for (int c = 0; c < 2; ++c)
#pragma unroll
        for (int t = 0; t < 16; ++t) {
            int kl = (hi ? 8 : 0) + ((t < 8) ? t : (t + 8));
            hjv[c][t] = hb[(size_t)(jbase + lr) * HID + c * 32 + kl];
        }

    // B fragments (w_d), hoisted
    v16h bfrag[2][4];
#pragma unroll
    for (int c = 0; c < 2; ++c)
#pragma unroll
        for (int ft = 0; ft < 4; ++ft)
#pragma unroll
            for (int t = 0; t < 16; ++t) {
                int kl = (hi ? 8 : 0) + ((t < 8) ? t : (t + 8));
                int k  = c * 32 + kl;
                bfrag[c][ft][t] = (_Float16)ps_w1[(size_t)(128 + k) * HID + ft * 16 + lr];
            }

    // s_j for this j tile: C-layout gives (VGPR r, lane) -> j = jbase + r + hi*8, f = ft*16+lr
    float sjv[4][8];
#pragma unroll
    for (int ft = 0; ft < 4; ++ft)
#pragma unroll
        for (int r = 0; r < 8; ++r) {
            int j = jbase + r + (hi ? 8 : 0);
            sjv[ft][r] = sjg[((size_t)b * NN + j) * HID + ft * 16 + lr];
        }

    float b1v[4], w2v[4];
#pragma unroll
    for (int ft = 0; ft < 4; ++ft) {
        b1v[ft] = ps_b1[ft * 16 + lr];
        w2v[ft] = ps_w2[ft * 16 + lr];
    }
    const float b2 = ps_b2[0];

    for (int i = 0; i < 16; ++i) {
        // A fragment: |h_i[k] - h_j[row][k]|, row = lane%16
        v16h afrag[2];
#pragma unroll
        for (int c = 0; c < 2; ++c)
#pragma unroll
            for (int t = 0; t < 16; ++t) {
                int kl = (hi ? 8 : 0) + ((t < 8) ? t : (t + 8));
                int k  = c * 32 + kl;
                afrag[c][t] = (_Float16)fabsf(lhi[i * HID + k] - hjv[c][t]);
            }

        v8f cacc[4];
#pragma unroll
        for (int ft = 0; ft < 4; ++ft) {
            v8f cz = {0.f, 0.f, 0.f, 0.f, 0.f, 0.f, 0.f, 0.f};
            cz = __builtin_amdgcn_wmma_f32_16x16x32_f16(false, afrag[0], false, bfrag[0][ft],
                                                        (short)0, cz, false, false);
            cz = __builtin_amdgcn_wmma_f32_16x16x32_f16(false, afrag[1], false, bfrag[1][ft],
                                                        (short)0, cz, false, false);
            cacc[ft] = cz;
        }

        // epilogue: + s_i + s_j + b1, relu, dot with w2 over 64 features
        float siv[4];
#pragma unroll
        for (int ft = 0; ft < 4; ++ft)
            siv[ft] = lsi[i * HID + ft * 16 + lr] + b1v[ft];

        float acc[8];
#pragma unroll
        for (int r = 0; r < 8; ++r) acc[r] = 0.f;
#pragma unroll
        for (int ft = 0; ft < 4; ++ft)
#pragma unroll
            for (int r = 0; r < 8; ++r) {
                float v = cacc[ft][r] + siv[ft] + sjv[ft][r];
                v = fmaxf(v, 0.f);
                acc[r] = fmaf(v, w2v[ft], acc[r]);
            }

        // reduce over the 16 lanes of each half (features live across lanes)
#pragma unroll
        for (int r = 0; r < 8; ++r) {
            float v = acc[r];
            v += __shfl_xor(v, 1, 32);
            v += __shfl_xor(v, 2, 32);
            v += __shfl_xor(v, 4, 32);
            v += __shfl_xor(v, 8, 32);
            acc[r] = v;
        }

        // lane lr (<8) holds result for r = lr; j = jbase + lr + hi*8
        float outv = acc[0];
#pragma unroll
        for (int r = 1; r < 8; ++r) outv = (lr == r) ? acc[r] : outv;
        if (lr < 8) {
            float v = tanhf(outv + b2);
            int j = jbase + lr + (hi ? 8 : 0);
            delta_raw[((size_t)b * NN + (ibase + i)) * NN + j] = v;
        }
    }
}

// ---------------------------------------------------------------------------
// Kernel 3: zero-diag, symmetrize, blend with a_static, relu, row-normalize
// ---------------------------------------------------------------------------
__global__ __launch_bounds__(256) void finalize_kernel(
    const float* __restrict__ a_static,
    const float* __restrict__ raw,         // B,N,N
    const float* __restrict__ raw_lambda,
    float* __restrict__ out)               // B*N*N floats + lam at the end
{
    const int i   = blockIdx.x % NN;
    const int b   = blockIdx.x / NN;
    const int tid = threadIdx.x;

    __shared__ float rowbuf[NN];
    __shared__ float psum[256];

    const float lam = 1.f / (1.f + expf(-raw_lambda[0]));

    const float* ri = raw + ((size_t)b * NN + i) * NN;     // raw[b,i,:]
    const float* rj = raw + (size_t)b * NN * NN + i;       // raw[b,:,i] (stride N)
    const float* as = a_static + (size_t)i * NN;

    float s = 0.f;
    for (int j = tid; j < NN; j += 256) {
        float d = (j == i) ? 0.f : 0.5f * (ri[j] + rj[(size_t)j * NN]);
        float v = as[j] + lam * d;
        v = fmaxf(v, 0.f);
        rowbuf[j] = v;
        s += v;
    }
    psum[tid] = s;
    __syncthreads();
    for (int off = 128; off > 0; off >>= 1) {
        if (tid < off) psum[tid] += psum[tid + off];
        __syncthreads();
    }
    const float inv = 1.f / fmaxf(psum[0], 1e-6f);
    float* o = out + ((size_t)b * NN + i) * NN;
    for (int j = tid; j < NN; j += 256) o[j] = rowbuf[j] * inv;

    if (b == 0 && i == 0 && tid == 0) out[(size_t)BB * NN * NN] = lam;
}

// ---------------------------------------------------------------------------
extern "C" void kernel_launch(void* const* d_in, const int* in_sizes, int n_in,
                              void* d_out, int out_size, void* d_ws, size_t ws_size,
                              hipStream_t stream) {
    const float* x_hist   = (const float*)d_in[0];
    const float* x_mark   = (const float*)d_in[1];
    const float* a_static = (const float*)d_in[2];
    const float* te_w1    = (const float*)d_in[3];
    const float* te_b1    = (const float*)d_in[4];
    const float* te_w2    = (const float*)d_in[5];
    const float* te_b2    = (const float*)d_in[6];
    const float* me_w1    = (const float*)d_in[7];
    const float* me_b1    = (const float*)d_in[8];
    const float* me_w2    = (const float*)d_in[9];
    const float* me_b2    = (const float*)d_in[10];
    const float* nf_w     = (const float*)d_in[11];
    const float* nf_b     = (const float*)d_in[12];
    const float* ps_w1    = (const float*)d_in[13];
    const float* ps_b1    = (const float*)d_in[14];
    const float* ps_w2    = (const float*)d_in[15];
    const float* ps_b2    = (const float*)d_in[16];
    const float* raw_lam  = (const float*)d_in[17];

    float* out = (float*)d_out;

    // workspace layout
    float* hg  = (float*)d_ws;                     // B*N*64
    float* sig = hg  + (size_t)BB * NN * HID;      // B*N*64
    float* sjg = sig + (size_t)BB * NN * HID;      // B*N*64
    float* raw = sjg + (size_t)BB * NN * HID;      // B*N*N

    node_feat_kernel<<<BB * NN, 64, 0, stream>>>(
        x_hist, x_mark, te_w1, te_b1, te_w2, te_b2,
        me_w1, me_b1, me_w2, me_b2, nf_w, nf_b, ps_w1,
        hg, sig, sjg);

    pair_wmma_kernel<<<dim3(NN / 16, NN / 16, BB), 32, 0, stream>>>(
        hg, sig, sjg, ps_w1, ps_b1, ps_w2, ps_b2, raw);

    finalize_kernel<<<BB * NN, 256, 0, stream>>>(a_static, raw, raw_lam, out);
}
